// Attention_84258668413378
// MI455X (gfx1250) — matrix operand order repaired and checked by reading, not compiled, not run on hardware
//
#include <hip/hip_runtime.h>
#include <hip/hip_bf16.h>

typedef __bf16 bf16;
typedef __attribute__((ext_vector_type(16))) __bf16 v16bf;
typedef __attribute__((ext_vector_type(8)))  __bf16 v8bf;
typedef __attribute__((ext_vector_type(8)))  float  v8f;
typedef __attribute__((ext_vector_type(4)))  int    i32x4;

#define LOG2E 1.4426950408889634f
#define SCALE_ 0.125f   // HD^-0.5, HD=64

#if defined(__AMDGCN__) && __has_builtin(__builtin_amdgcn_global_load_async_to_lds_b128)
#define HAVE_ASYNC_B128 1
#else
#define HAVE_ASYNC_B128 0
#endif

// ---------------------------------------------------------------- helpers

__device__ __forceinline__ v8f wmma_bf16(v16bf a, v16bf b, v8f c) {
  return __builtin_amdgcn_wmma_f32_16x16x32_bf16(false, a, false, b, (short)0, c,
                                                 false, false);
}

__device__ __forceinline__ v16bf cat8(v8bf lo, v8bf hi) {
  return __builtin_shufflevector(lo, hi, 0, 1, 2, 3, 4, 5, 6, 7,
                                 8, 9, 10, 11, 12, 13, 14, 15);
}

#if HAVE_ASYNC_B128
// async 16B global -> LDS copy (ASYNCcnt path); builtin wants typed AS pointers
__device__ __forceinline__ void async_copy16(const v8bf* g, bf16* l) {
  __builtin_amdgcn_global_load_async_to_lds_b128(
      (__attribute__((address_space(1))) i32x4*)(g),
      (__attribute__((address_space(3))) i32x4*)(l), 0, 0);
}
__device__ __forceinline__ void wait_async0() {
#if __has_builtin(__builtin_amdgcn_s_wait_asynccnt)
  __builtin_amdgcn_s_wait_asynccnt(0);
#else
  asm volatile("s_wait_asynccnt 0x0" ::: "memory");
#endif
}
#endif

// A-operand (16x32, MxK): lane row = lane&15; K base = (lane>>4)*8;
// elems 0..7 -> K kb..kb+7, elems 8..15 -> K kb+16..kb+23.  Two ds_load_b128.
__device__ __forceinline__ v16bf load_a16(const bf16* smem, int row_base, int ld) {
  const int lane = threadIdx.x & 31;
  const v8bf* p =
      (const v8bf*)(smem + (row_base + (lane & 15)) * ld + ((lane >> 4) << 3));
  return cat8(p[0], p[2]);
}

// B-operand (32x16, KxN): lane col = lane&15; K base = (lane>>4)*8;
// elems 0..7 -> K kb..kb+7, elems 8..15 -> K kb+16..kb+23 (source [N][K]).
__device__ __forceinline__ v16bf load_b16(const bf16* smem, int col_base, int ld) {
  const int lane = threadIdx.x & 31;
  const v8bf* p =
      (const v8bf*)(smem + (col_base + (lane & 15)) * ld + ((lane >> 4) << 3));
  return cat8(p[0], p[2]);
}

// fp32x8 -> packed bf16x8 LDS store (single ds_store_b128)
__device__ __forceinline__ void store_v8(bf16* dst, float4 a, float4 b) {
  v8bf t;
  t[0] = (bf16)a.x; t[1] = (bf16)a.y; t[2] = (bf16)a.z; t[3] = (bf16)a.w;
  t[4] = (bf16)b.x; t[5] = (bf16)b.y; t[6] = (bf16)b.z; t[7] = (bf16)b.w;
  *(v8bf*)dst = t;
}

// ---------------------------------------------------------------- kernel 1: QKV

// qkv = x[8192,512] @ w_qkv[1536,512]^T ; scatter q(scaled)/k/v -> [B,H,N,64] bf16
__global__ __launch_bounds__(256) void qkv_gemm(const float* __restrict__ x,
                                                const float* __restrict__ w_qkv,
                                                bf16* __restrict__ q_ws,
                                                bf16* __restrict__ k_ws,
                                                bf16* __restrict__ v_ws) {
  __shared__ bf16 As[64 * 40];
  __shared__ bf16 Bs[128 * 40];
  const int mb = blockIdx.x << 6;       // [0,8192)
  const int cb = blockIdx.y << 7;       // [0,1536)
  const int tid = threadIdx.x;
  const int wave = tid >> 5, lane = tid & 31;
  const int mtile = (wave >> 1) << 4;
  const int nbase = (wave & 1) << 6;
  const int sr = tid >> 2;              // staging row 0..63
  const int sc = (tid & 3) << 3;        // staging col 0,8,16,24
  v8f acc[4] = {};
  for (int kb = 0; kb < 512; kb += 32) {
    float4 f0 = *(const float4*)&x[(mb + sr) * 512 + kb + sc];
    float4 f1 = *(const float4*)&x[(mb + sr) * 512 + kb + sc + 4];
    float4 g0 = *(const float4*)&w_qkv[(cb + sr) * 512 + kb + sc];
    float4 g1 = *(const float4*)&w_qkv[(cb + sr) * 512 + kb + sc + 4];
    float4 g2 = *(const float4*)&w_qkv[(cb + sr + 64) * 512 + kb + sc];
    float4 g3 = *(const float4*)&w_qkv[(cb + sr + 64) * 512 + kb + sc + 4];
    store_v8(&As[sr * 40 + sc], f0, f1);
    store_v8(&Bs[sr * 40 + sc], g0, g1);
    store_v8(&Bs[(sr + 64) * 40 + sc], g2, g3);
    __syncthreads();
    v16bf a  = load_a16(As, mtile, 40);
    v16bf b0 = load_b16(Bs, nbase +  0, 40);
    v16bf b1 = load_b16(Bs, nbase + 16, 40);
    v16bf b2 = load_b16(Bs, nbase + 32, 40);
    v16bf b3 = load_b16(Bs, nbase + 48, 40);
    acc[0] = wmma_bf16(a, b0, acc[0]);
    acc[1] = wmma_bf16(a, b1, acc[1]);
    acc[2] = wmma_bf16(a, b2, acc[2]);
    acc[3] = wmma_bf16(a, b3, acc[3]);
    __syncthreads();
  }
  const int col = lane & 15;
  const int rbase = mtile + ((lane >> 4) << 3);
  const int bidx = mb >> 11;
  const int n0 = (mb & 2047) + rbase;
#pragma unroll
  for (int t = 0; t < 4; ++t) {
    const int c = cb + nbase + (t << 4) + col;
    const int sect = c >> 9;                   // 0=q 1=k 2=v (tile-uniform)
    const int cc = c & 511;
    const int h = cc >> 6, d = cc & 63;
    bf16* dst = (sect == 0) ? q_ws : (sect == 1 ? k_ws : v_ws);
    const float sca = (sect == 0) ? SCALE_ : 1.0f;
    bf16* base = dst + (((bidx << 3) + h) * 2048 + n0) * 64 + d;
#pragma unroll
    for (int i = 0; i < 8; ++i)
      base[i * 64] = (bf16)(acc[t][i] * sca);
  }
}

// --------------------------------------------------------- kernel 2: flash attn

// grid (N/64, B*H); 4 waves/block, each wave owns 16 query rows.
// Software-pipelined: chunk c+1's K (async->LDS) and V (global->regs) issued
// right after the single per-iteration barrier, overlapping chunk c's compute.
__global__ __launch_bounds__(128) void flash_attn(const bf16* __restrict__ q_ws,
                                                  const bf16* __restrict__ k_ws,
                                                  const bf16* __restrict__ v_ws,
                                                  const float* __restrict__ table,
                                                  const int* __restrict__ rel_idx,
                                                  const unsigned char* __restrict__ mask,
                                                  bf16* __restrict__ o_ws) {
  __shared__ bf16 Ks[2][32 * 64];     // K chunk [key][d], double-buffered
  __shared__ bf16 Vt[2][64 * 40];     // V chunk transposed [d][key], 2 bufs
  __shared__ bf16 Ps[4][16 * 32];     // wave-private P staging [row][key]
  const int bh = blockIdx.y;
  const int bb = bh >> 3, h = bh & 7;
  const int tid = threadIdx.x, wave = tid >> 5, lane = tid & 31;
  const int laneHi = lane >> 4, col = lane & 15;
  const int qtile = (blockIdx.x << 6) + (wave << 4);
  // Q fragments (16x64 bf16 -> two 16x32 A operands), 4x16B loads per lane
  v16bf qa0, qa1;
  {
    const v8bf* p = (const v8bf*)(q_ws + (bh * 2048 + qtile) * 64 +
                                  (lane & 15) * 64 + (laneHi << 3));
    qa0 = cat8(p[0], p[2]);
    qa1 = cat8(p[4], p[6]);
  }
  float mrow[8], lrow[8];
#pragma unroll
  for (int i = 0; i < 8; ++i) { mrow[i] = -3.0e38f; lrow[i] = 0.f; }
  v8f o[4] = {};
  const v8bf* kg8 = (const v8bf*)(k_ws + bh * 2048 * 64);
  const v8bf* vg8 = (const v8bf*)(v_ws + bh * 2048 * 64);
  const int vkey = tid >> 3;            // 0..15
  const int vd = (tid & 7) << 3;        // 0..56

  // ---- prologue: stage chunk 0 ----
#if HAVE_ASYNC_B128
  async_copy16(&kg8[tid],       &Ks[0][tid << 3]);
  async_copy16(&kg8[128 + tid], &Ks[0][(tid + 128) << 3]);
#else
  {
    v8bf ka = kg8[tid], kc = kg8[128 + tid];
    *(v8bf*)&Ks[0][tid << 3]         = ka;
    *(v8bf*)&Ks[0][(tid + 128) << 3] = kc;
  }
#endif
  {
    v8bf va = vg8[tid], vb = vg8[128 + tid];
#pragma unroll
    for (int j = 0; j < 8; ++j) {
      Vt[0][(vd + j) * 40 + vkey]      = va[j];
      Vt[0][(vd + j) * 40 + vkey + 16] = vb[j];
    }
  }

  for (int kv = 0; kv < 2048; kv += 32) {
    const int buf = (kv >> 5) & 1;
    const int nbuf = buf ^ 1;
    const bool more = (kv + 32) < 2048;
#if HAVE_ASYNC_B128
    wait_async0();                      // this wave's chunk-c K is in LDS
#endif
    __syncthreads();                    // all waves staged c, done computing c-1
    // ---- issue chunk c+1 staging (overlaps compute below) ----
    v8bf va = {}, vb = {};
#if HAVE_ASYNC_B128
    if (more) {
      const int nb8 = (kv + 32) * 8;
      async_copy16(&kg8[nb8 + tid],       &Ks[nbuf][tid << 3]);
      async_copy16(&kg8[nb8 + 128 + tid], &Ks[nbuf][(tid + 128) << 3]);
      va = vg8[nb8 + tid];
      vb = vg8[nb8 + 128 + tid];
    }
#else
    v8bf ka = {}, kc = {};
    if (more) {
      const int nb8 = (kv + 32) * 8;
      ka = kg8[nb8 + tid];
      kc = kg8[nb8 + 128 + tid];
      va = vg8[nb8 + tid];
      vb = vg8[nb8 + 128 + tid];
    }
#endif
    // ---- S = q @ k^T : load all 4 B-frags, then 4 WMMAs ----
    const bf16* K = Ks[buf];
    v16bf b00 = load_b16(K,      0, 64);
    v16bf b01 = load_b16(K + 32, 0, 64);
    v16bf b10 = load_b16(K,      16, 64);
    v16bf b11 = load_b16(K + 32, 16, 64);
    v8f s0 = {}, s1 = {};
    s0 = wmma_bf16(qa0, b00, s0);
    s1 = wmma_bf16(qa0, b10, s1);
    s0 = wmma_bf16(qa1, b01, s0);
    s1 = wmma_bf16(qa1, b11, s1);
    // ---- relative-position bias gather (L2-resident) + key mask ----
    const int k0 = kv + col, k1 = kv + 16 + col;
    const float mk0 = mask[bb * 2048 + k0] ? 0.f : -3.0e38f;
    const float mk1 = mask[bb * 2048 + k1] ? 0.f : -3.0e38f;
#pragma unroll
    for (int i = 0; i < 8; ++i) {
      const int rq = qtile + i + (laneHi << 3);
      s0[i] = fmaxf(s0[i] + table[rel_idx[rq * 2048 + k0] * 8 + h] + mk0, -3.0e38f);
      s1[i] = fmaxf(s1[i] + table[rel_idx[rq * 2048 + k1] * 8 + h] + mk1, -3.0e38f);
    }
    // ---- online softmax: reductions across 16-lane half-wave groups ----
    bf16* pw = &Ps[wave][0];
#pragma unroll
    for (int i = 0; i < 8; ++i) {
      float v = fmaxf(s0[i], s1[i]);
      v = fmaxf(v, __shfl_xor(v, 1, 16));
      v = fmaxf(v, __shfl_xor(v, 2, 16));
      v = fmaxf(v, __shfl_xor(v, 4, 16));
      v = fmaxf(v, __shfl_xor(v, 8, 16));
      const float mnew  = fmaxf(mrow[i], v);
      const float alpha = exp2f((mrow[i] - mnew) * LOG2E);
      mrow[i] = mnew;
      const float p0 = exp2f((s0[i] - mnew) * LOG2E);
      const float p1 = exp2f((s1[i] - mnew) * LOG2E);
      float rs = p0 + p1;
      rs += __shfl_xor(rs, 1, 16);
      rs += __shfl_xor(rs, 2, 16);
      rs += __shfl_xor(rs, 4, 16);
      rs += __shfl_xor(rs, 8, 16);
      lrow[i] = lrow[i] * alpha + rs;
      const int r = i + (laneHi << 3);
      pw[r * 32 + col]      = (bf16)p0;   // wave-private; DS ops are in-order
      pw[r * 32 + 16 + col] = (bf16)p1;
#pragma unroll
      for (int t = 0; t < 4; ++t) o[t][i] = o[t][i] * alpha;
    }
    // ---- O += P @ V : all frag loads first, then 4 WMMAs ----
    const bf16* V = Vt[buf];
    v16bf pa  = load_a16(pw, 0, 32);
    v16bf vb0 = load_b16(V,  0, 40);
    v16bf vb1 = load_b16(V, 16, 40);
    v16bf vb2 = load_b16(V, 32, 40);
    v16bf vb3 = load_b16(V, 48, 40);
    o[0] = wmma_bf16(pa, vb0, o[0]);
    o[1] = wmma_bf16(pa, vb1, o[1]);
    o[2] = wmma_bf16(pa, vb2, o[2]);
    o[3] = wmma_bf16(pa, vb3, o[3]);
    // ---- stage chunk c+1 into the other buffers (pre-barrier of iter c+1) ----
    if (more) {
#if !HAVE_ASYNC_B128
      *(v8bf*)&Ks[nbuf][tid << 3]         = ka;
      *(v8bf*)&Ks[nbuf][(tid + 128) << 3] = kc;
#endif
#pragma unroll
      for (int j = 0; j < 8; ++j) {
        Vt[nbuf][(vd + j) * 40 + vkey]      = va[j];
        Vt[nbuf][(vd + j) * 40 + vkey + 16] = vb[j];
      }
    }
  }
  // epilogue: normalize, store bf16 in [B,N, h*64+d] layout (proj input)
#pragma unroll
  for (int i = 0; i < 8; ++i) {
    const float inv = 1.0f / lrow[i];
    const int n = qtile + i + (laneHi << 3);
    bf16* op = o_ws + (((bb << 11) + n) << 9) + (h << 6);
#pragma unroll
    for (int t = 0; t < 4; ++t)
      op[(t << 4) + col] = (bf16)(o[t][i] * inv);
  }
}

// --------------------------------------------------------- kernel 3: projection

__global__ __launch_bounds__(256) void proj_gemm(const bf16* __restrict__ o_ws,
                                                 const float* __restrict__ w_proj,
                                                 const float* __restrict__ b_proj,
                                                 float* __restrict__ out) {
  __shared__ bf16 As[64 * 40];
  __shared__ bf16 Bs[128 * 40];
  const int mb = blockIdx.x << 6;
  const int cb = blockIdx.y << 7;
  const int tid = threadIdx.x;
  const int wave = tid >> 5, lane = tid & 31;
  const int mtile = (wave >> 1) << 4;
  const int nbase = (wave & 1) << 6;
  const int sr = tid >> 2;
  const int sc = (tid & 3) << 3;
  v8f acc[4] = {};
  for (int kb = 0; kb < 512; kb += 32) {
    v8bf  a8 = *(const v8bf*)&o_ws[(mb + sr) * 512 + kb + sc];  // already bf16
    float4 g0 = *(const float4*)&w_proj[(cb + sr) * 512 + kb + sc];
    float4 g1 = *(const float4*)&w_proj[(cb + sr) * 512 + kb + sc + 4];
    float4 g2 = *(const float4*)&w_proj[(cb + sr + 64) * 512 + kb + sc];
    float4 g3 = *(const float4*)&w_proj[(cb + sr + 64) * 512 + kb + sc + 4];
    *(v8bf*)&As[sr * 40 + sc] = a8;
    store_v8(&Bs[sr * 40 + sc], g0, g1);
    store_v8(&Bs[(sr + 64) * 40 + sc], g2, g3);
    __syncthreads();
    v16bf a  = load_a16(As, mtile, 40);
    v16bf b0 = load_b16(Bs, nbase +  0, 40);
    v16bf b1 = load_b16(Bs, nbase + 16, 40);
    v16bf b2 = load_b16(Bs, nbase + 32, 40);
    v16bf b3 = load_b16(Bs, nbase + 48, 40);
    acc[0] = wmma_bf16(a, b0, acc[0]);
    acc[1] = wmma_bf16(a, b1, acc[1]);
    acc[2] = wmma_bf16(a, b2, acc[2]);
    acc[3] = wmma_bf16(a, b3, acc[3]);
    __syncthreads();
  }
  const int col = lane & 15;
  const int rbase = mtile + ((lane >> 4) << 3);
#pragma unroll
  for (int t = 0; t < 4; ++t) {
    const int c = cb + nbase + (t << 4) + col;
    const float bias = b_proj[c];
#pragma unroll
    for (int i = 0; i < 8; ++i)
      out[(mb + rbase + i) * 512 + c] = acc[t][i] + bias;
  }
}

// ------------------------------------------------------------------- launcher

extern "C" void kernel_launch(void* const* d_in, const int* in_sizes, int n_in,
                              void* d_out, int out_size, void* d_ws, size_t ws_size,
                              hipStream_t stream) {
  // setup_inputs order: x, w_qkv, rel_bias_table, w_proj, b_proj, mask, rel_idx
  const float* x       = (const float*)d_in[0];
  const float* w_qkv   = (const float*)d_in[1];
  const float* table   = (const float*)d_in[2];
  const float* w_proj  = (const float*)d_in[3];
  const float* b_proj  = (const float*)d_in[4];
  const unsigned char* mask = (const unsigned char*)d_in[5];  // bool array
  const int*   rel_idx = (const int*)d_in[6];
  float* out = (float*)d_out;

  char* ws = (char*)d_ws;
  bf16* q_ws = (bf16*)(ws);                 // 8 MiB
  bf16* k_ws = (bf16*)(ws + 8388608);       // 8 MiB
  bf16* v_ws = (bf16*)(ws + 16777216);      // 8 MiB
  bf16* o_ws = (bf16*)(ws + 25165824);      // 8 MiB  (total 32 MiB)

  hipLaunchKernelGGL(qkv_gemm, dim3(128, 12), dim3(256), 0, stream,
                     x, w_qkv, q_ws, k_ws, v_ws);
  hipLaunchKernelGGL(flash_attn, dim3(32, 32), dim3(128), 0, stream,
                     q_ws, k_ws, v_ws, table, rel_idx, mask, o_ws);
  hipLaunchKernelGGL(proj_gemm, dim3(128, 4), dim3(256), 0, stream,
                     o_ws, w_proj, b_proj, out);
}
